// M_25005299597627
// MI455X (gfx1250) — matrix operand order repaired and checked by reading, not compiled, not run on hardware
//
#include <hip/hip_runtime.h>
#include <stdint.h>

#define NUM_HEADS 12
#define NUM_KV_HEADS 2
#define HEAD_SIZE 128
#define GROUP (NUM_HEADS / NUM_KV_HEADS)
#define Q_STRIDE (NUM_HEADS * HEAD_SIZE)       // 1536 floats per token
#define KV_STRIDE (NUM_KV_HEADS * HEAD_SIZE)   // 256 floats per token
#define SCALE 0.08838834764831845f

// Padded LDS row strides (ushorts) -> bank-conflict-free with immediate offsets.
#define KSTR 136    // K tile rows: 128 dims + 8 pad  (272B/row, 68 dwords)
#define VSTR 40     // V^T rows: 32 keys + 8 pad      (80B/row, 20 dwords)
#define PSTR 40     // P pad rows: 32 + 8 pad

typedef __attribute__((ext_vector_type(16))) __bf16 v16bf;
typedef __attribute__((ext_vector_type(8)))  float  v8f;
typedef __attribute__((ext_vector_type(8)))  unsigned short us8;
typedef __attribute__((ext_vector_type(4)))  float  f4;

__device__ __forceinline__ unsigned short f2bf(float f) {
    return __builtin_bit_cast(unsigned short, (__bf16)f);
}
__device__ __forceinline__ unsigned pack2bf(float a, float b) {
    return (unsigned)f2bf(a) | ((unsigned)f2bf(b) << 16);
}

// Build a 16-element bf16 fragment from two 8-element (16B) LDS chunks.
__device__ __forceinline__ v16bf ld16_2(const unsigned short* p0, const unsigned short* p1) {
    union { v16bf v; us8 h[2]; } u;
    u.h[0] = *(const us8*)p0;
    u.h[1] = *(const us8*)p1;
    return u.v;
}
__device__ __forceinline__ v16bf ld16_c(const unsigned short* p) { return ld16_2(p, p + 8); }

__global__ __launch_bounds__(256) void fa_gqa_causal(const float* __restrict__ qg,
                                                     const float* __restrict__ kg,
                                                     const float* __restrict__ vg,
                                                     float* __restrict__ outg,
                                                     int S) {
    __shared__ __align__(16) unsigned short lds_k[2][32 * KSTR];    // [key][dim] bf16
    __shared__ __align__(16) unsigned short lds_vt[2][128 * VSTR];  // [dim][key] bf16
    __shared__ __align__(16) unsigned short lds_p[8][16 * PSTR];    // per-wave [row][k]

    const int tid  = threadIdx.x;
    const int w    = tid >> 5;
    const int lane = tid & 31;
    const int lm   = lane & 15;
    const int hi   = lane >> 4;
    const int bx   = blockIdx.x;
    const int h    = blockIdx.y;
    const int kvh  = h / GROUP;

    const int q0     = bx * 128 + w * 16;
    const int q_hi_s = __builtin_amdgcn_readfirstlane(q0 + 15);   // scalar -> EXEC untouched

    // ---- Q tile -> 4 A-fragments (pre-scaled by 1/sqrt(d)), bf16 ----
    const float* qrow = qg + (size_t)(q0 + lm) * Q_STRIDE + h * HEAD_SIZE;
    v16bf qa[4];
#pragma unroll
    for (int t = 0; t < 4; ++t) {
        const int base = t * 32 + 16 * hi;
        f4 a0 = *(const f4*)(qrow + base);
        f4 a1 = *(const f4*)(qrow + base + 4);
        f4 b0 = *(const f4*)(qrow + base + 8);
        f4 b1 = *(const f4*)(qrow + base + 12);
        union { v16bf v; unsigned u[8]; } u;
        u.u[0] = pack2bf(a0.x * SCALE, a0.y * SCALE);
        u.u[1] = pack2bf(a0.z * SCALE, a0.w * SCALE);
        u.u[2] = pack2bf(a1.x * SCALE, a1.y * SCALE);
        u.u[3] = pack2bf(a1.z * SCALE, a1.w * SCALE);
        u.u[4] = pack2bf(b0.x * SCALE, b0.y * SCALE);
        u.u[5] = pack2bf(b0.z * SCALE, b0.w * SCALE);
        u.u[6] = pack2bf(b1.x * SCALE, b1.y * SCALE);
        u.u[7] = pack2bf(b1.z * SCALE, b1.w * SCALE);
        qa[t] = u.v;
    }

    // ---- Flash state ----
    v8f acc[8];
#pragma unroll
    for (int f = 0; f < 8; ++f) acc[f] = (v8f){0,0,0,0,0,0,0,0};
    float mrow[8], lrow[8];
#pragma unroll
    for (int i = 0; i < 8; ++i) { mrow[i] = -3.0e38f; lrow[i] = 0.0f; }

    // Per-lane LDS bases (all loop-body offsets become immediates).
    const int kb_lo = lm * KSTR + 16 * hi;         // K rows lm / 16+lm
    const int kb_hi = (16 + lm) * KSTR + 16 * hi;
    const int vt_b  = lm * VSTR + 16 * hi;         // + f*16*VSTR immediate
    const int p_rd  = lm * PSTR + 16 * hi;         // A-layout reload base

    // K staging: lane owns dim-column col = lane*4, rows w + 8i (coalesced).
    const int kc = lane * 4;
    const float* kcol = kg + kvh * HEAD_SIZE + kc;
    // V staging: lane owns key row 'lane', dims w*16 .. +15 (conflict-free V^T stores).
    const float* vrow_base = vg + kvh * HEAD_SIZE + w * 16;

    f4 kr[4], vr[4];
    auto stage_load = [&](int k0) {
#pragma unroll
        for (int i = 0; i < 4; ++i)
            kr[i] = *(const f4*)(kcol + (size_t)(k0 + w + 8 * i) * KV_STRIDE);
        const float* vrow = vrow_base + (size_t)(k0 + lane) * KV_STRIDE;
#pragma unroll
        for (int i = 0; i < 4; ++i)
            vr[i] = *(const f4*)(vrow + i * 4);
    };
    auto stage_store = [&](int buf) {
#pragma unroll
        for (int i = 0; i < 4; ++i) {
            const int row = w + 8 * i;
            unsigned* dk = (unsigned*)&lds_k[buf][row * KSTR + kc];
            dk[0] = pack2bf(kr[i].x, kr[i].y);
            dk[1] = pack2bf(kr[i].z, kr[i].w);
            unsigned short* dv = &lds_vt[buf][(w * 16 + i * 4) * VSTR + lane];
            dv[0 * VSTR] = f2bf(vr[i].x);
            dv[1 * VSTR] = f2bf(vr[i].y);
            dv[2 * VSTR] = f2bf(vr[i].z);
            dv[3 * VSTR] = f2bf(vr[i].w);
        }
    };

    const int nkb = (bx + 1) * 4;

    stage_load(0);
    stage_store(0);
    __syncthreads();

    for (int kb = 0; kb < nkb; ++kb) {
        const int k0  = kb * 32;
        const int cur = kb & 1;
        const bool have_next = (kb + 1) < nkb;       // block-uniform

        if (have_next) stage_load(k0 + 32);          // HBM fill overlaps WMMA below

        if (k0 <= q_hi_s) {
            const unsigned short* kbuf = &lds_k[cur][0];
            const unsigned short* vbuf = &lds_vt[cur][0];

            // ---- S = Q*K^T: 4 independent depth-2 WMMA chains ----
            v8f c0a = (v8f){0,0,0,0,0,0,0,0}, c0b = (v8f){0,0,0,0,0,0,0,0};
            v8f c1a = (v8f){0,0,0,0,0,0,0,0}, c1b = (v8f){0,0,0,0,0,0,0,0};
#pragma unroll
            for (int t = 0; t < 2; ++t) {
                v16bf k00 = ld16_c(kbuf + kb_lo + t * 32);
                v16bf k10 = ld16_c(kbuf + kb_hi + t * 32);
                v16bf k01 = ld16_c(kbuf + kb_lo + (t + 2) * 32);
                v16bf k11 = ld16_c(kbuf + kb_hi + (t + 2) * 32);
                c0a = __builtin_amdgcn_wmma_f32_16x16x32_bf16(false, qa[t],     false, k00, (short)0, c0a, false, false);
                c1a = __builtin_amdgcn_wmma_f32_16x16x32_bf16(false, qa[t],     false, k10, (short)0, c1a, false, false);
                c0b = __builtin_amdgcn_wmma_f32_16x16x32_bf16(false, qa[t + 2], false, k01, (short)0, c0b, false, false);
                c1b = __builtin_amdgcn_wmma_f32_16x16x32_bf16(false, qa[t + 2], false, k11, (short)0, c1b, false, false);
            }
            v8f c0 = c0a + c0b;
            v8f c1 = c1a + c1b;

            // ---- Online softmax (scale folded into Q) ----
            const bool need_mask = (k0 + 31 > q0);
            float corr[8];
#pragma unroll
            for (int vv = 0; vv < 8; ++vv) {
                const int row = q0 + vv + 8 * hi;
                float s0 = c0[vv];
                float s1 = c1[vv];
                if (need_mask) {
                    if (k0 + lm      > row) s0 = -3.0e38f;
                    if (k0 + 16 + lm > row) s1 = -3.0e38f;
                }
                float mx = fmaxf(s0, s1);
                mx = fmaxf(mx, __shfl_xor(mx, 1, 32));
                mx = fmaxf(mx, __shfl_xor(mx, 2, 32));
                mx = fmaxf(mx, __shfl_xor(mx, 4, 32));
                mx = fmaxf(mx, __shfl_xor(mx, 8, 32));
                const float mnew = fmaxf(mrow[vv], mx);
                const float p0 = __expf(s0 - mnew);
                const float p1 = __expf(s1 - mnew);
                float ps = p0 + p1;
                ps += __shfl_xor(ps, 1, 32);
                ps += __shfl_xor(ps, 2, 32);
                ps += __shfl_xor(ps, 4, 32);
                ps += __shfl_xor(ps, 8, 32);
                const float cr = __expf(mrow[vv] - mnew);
                lrow[vv] = lrow[vv] * cr + ps;
                mrow[vv] = mnew;
                corr[vv] = cr;
                unsigned short* pw = &lds_p[w][(vv + 8 * hi) * PSTR + lm];
                pw[0]  = f2bf(p0);
                pw[16] = f2bf(p1);
            }

            // Per-wave LDS round-trip: wait for DS stores, then reload in A-layout.
            asm volatile("s_wait_dscnt 0" ::: "memory");
            v16bf pa = ld16_2(&lds_p[w][p_rd], &lds_p[w][p_rd + 8]);

            // ---- O = diag(corr)*O + P*V : 8 independent WMMA chains ----
#pragma unroll
            for (int f = 0; f < 8; ++f) {
#pragma unroll
                for (int vv = 0; vv < 8; ++vv) acc[f][vv] *= corr[vv];
                v16bf vb = ld16_c(vbuf + vt_b + f * 16 * VSTR);
                acc[f] = __builtin_amdgcn_wmma_f32_16x16x32_bf16(false, pa, false, vb,
                                                                 (short)0, acc[f], false, false);
            }
        }

        if (have_next) stage_store(cur ^ 1);
        __syncthreads();
    }

    // ---- Epilogue: normalize, coalesced fp32 stores ----
    float inv[8];
#pragma unroll
    for (int vv = 0; vv < 8; ++vv) inv[vv] = 1.0f / lrow[vv];
#pragma unroll
    for (int f = 0; f < 8; ++f) {
#pragma unroll
        for (int vv = 0; vv < 8; ++vv) {
            const int row = q0 + vv + 8 * hi;
            outg[(size_t)row * Q_STRIDE + h * HEAD_SIZE + f * 16 + lm] = acc[f][vv] * inv[vv];
        }
    }
}

extern "C" void kernel_launch(void* const* d_in, const int* in_sizes, int n_in,
                              void* d_out, int out_size, void* d_ws, size_t ws_size,
                              hipStream_t stream) {
    const float* q = (const float*)d_in[0];
    const float* k = (const float*)d_in[1];
    const float* v = (const float*)d_in[2];
    float* out = (float*)d_out;
    const int S = in_sizes[0] / Q_STRIDE;       // 4096
    dim3 grid(S / 128, NUM_HEADS);
    dim3 block(256);
    fa_gqa_causal<<<grid, block, 0, stream>>>(q, k, v, out, S);
}
